// NTPLayer_53420803227697
// MI455X (gfx1250) — matrix operand order repaired and checked by reading, not compiled, not run on hardware
//
#include <hip/hip_runtime.h>
#include <math.h>

// ---------------------------------------------------------------------------
// Fused graph-transformer layer for MI455X (gfx1250, wave32, WMMA).
// One workgroup (8 wave32) per node; all GEMMs via v_wmma_f32_16x16x32_f16.
// Round 2: bank-conflict-free padded LDS pitches (dynamic 71KB LDS),
// B-fragment reuse (2 M-tiles per B load), vectorized 128-bit gather.
// ---------------------------------------------------------------------------

typedef __attribute__((ext_vector_type(16))) _Float16 v16h;
typedef __attribute__((ext_vector_type(8)))  _Float16 v8h;
typedef __attribute__((ext_vector_type(8)))  float    v8f;

#define NODES 8000
#define DIN   128
#define DOUT  256
#define HEADS 4
#define DH    64
#define MAXD  32
#define NEGV  (-1e30f)

// padded LDS pitches (in f16 elements) -> row stride 528B / 80B, both
// == 16 mod 64B so 16 lanes hit 16 distinct bank groups (2-req minimum).
#define PA 264   // [32][PA] activation tiles
#define PV 40    // [256][PV] transposed V
#define PT 40    // [4][32][PT] attention weights

// LDS pool offsets (bytes), all 32B aligned; total 71168B dynamic LDS.
#define OFF_P0 0          // A0 f16[32][PA] -> S f32[4][32][32] -> Val f32[32][128]
#define OFF_P1 16896      // EX f16[32][PA] -> OUTF f16[32][PA]
#define OFF_P2 33792      // Q  f16[32][PA] -> Vt f16[256][PV]
#define OFF_P3 54272      // K  f16[32][PA] -> ATT f16[4][32][PT] + logits/alpha
#define OFF_LOGITS (OFF_P3 + 10240)
#define OFF_ALPHA  (OFF_P3 + 10752)
#define SMEM_TOTAL 71168

__device__ __forceinline__ float gelu_exact(float x) {
  return 0.5f * x * (1.0f + erff(x * 0.70710678118654752f));
}

// A-fragment (16x32 f16, M x K). Lanes 0-15: row M=lane, K 0-7 / 16-23;
// lanes 16-31: row M=lane-16, K 8-15 / 24-31.  Two 16B loads per lane.
__device__ __forceinline__ v16h load_a16(const _Float16* base, int pitch, int lane) {
  const int row = lane & 15;
  const int kr  = (lane >> 4) << 3;
  const _Float16* p = base + row * pitch + kr;
  v8h lo = *(const v8h*)(p);
  v8h hi = *(const v8h*)(p + 16);
  v16h r;
#pragma unroll
  for (int i = 0; i < 8; ++i) { r[i] = lo[i]; r[8 + i] = hi[i]; }
  return r;
}

// B-fragment (32x16 f16, K x N) from B^T stored row-major (W[n][k]):
// lanes 0-15 col N=lane hold K 0-7 / 16-23; lanes 16-31 col N=lane-16, K 8-15 / 24-31.
// Two 16B loads per lane (pitch may be odd multiple of 16B).
__device__ __forceinline__ v16h load_b16(const _Float16* base, int pitch, int lane) {
  const int col = lane & 15;
  const int kb  = (lane >> 4) << 3;
  const _Float16* p = base + col * pitch + kb;
  v8h lo = *(const v8h*)(p);
  v8h hi = *(const v8h*)(p + 16);
  v16h r;
#pragma unroll
  for (int i = 0; i < 8; ++i) { r[i] = lo[i]; r[8 + i] = hi[i]; }
  return r;
}

__device__ __forceinline__ v8f wmma_f16(v16h a, v16h b, v8f c) {
  return __builtin_amdgcn_wmma_f32_16x16x32_f16(false, a, false, b, (short)0, c,
                                                false, false);
}

// --------------------------- small helper kernels ---------------------------

__global__ void cvt_f32_f16(const float* __restrict__ in, _Float16* __restrict__ out, int n) {
  int i = blockIdx.x * blockDim.x + threadIdx.x;
  if (i < n) out[i] = (_Float16)in[i];
}

__global__ void zero_f32(float* __restrict__ p, int n) {
  int i = blockIdx.x * blockDim.x + threadIdx.x;
  if (i < n) p[i] = 0.0f;
}

__global__ void zero_i32(int* __restrict__ p, int n) {
  int i = blockIdx.x * blockDim.x + threadIdx.x;
  if (i < n) p[i] = 0;
}

// e0 sorted; pos[i] = i - start[e0[i]].  Derive start[] and deg[].
__global__ void prep_edges(const int* __restrict__ e0, const int* __restrict__ pos,
                           int* __restrict__ start, int* __restrict__ deg, int m) {
  int i = blockIdx.x * blockDim.x + threadIdx.x;
  if (i >= m) return;
  const int v = e0[i];
  if (pos[i] == 0) start[v] = i;
  if (i == m - 1 || e0[i + 1] != v) deg[v] = pos[i] + 1;
}

// ------------------------------ main kernel --------------------------------

__global__ __launch_bounds__(256)
void node_attn_kernel(const _Float16* __restrict__ xh,
                      const _Float16* __restrict__ WcH, const _Float16* __restrict__ WqH,
                      const _Float16* __restrict__ WkH, const _Float16* __restrict__ WvH,
                      const _Float16* __restrict__ WoH,
                      const float* __restrict__ bc, const float* __restrict__ bq,
                      const float* __restrict__ bk, const float* __restrict__ bv,
                      const float* __restrict__ bo,
                      const int* __restrict__ e1, const int* __restrict__ start,
                      const int* __restrict__ deg_arr, float* __restrict__ out) {
  extern __shared__ __align__(32) unsigned char smem[];
  _Float16* A0     = (_Float16*)(smem + OFF_P0);
  float*    S      = (float*)(smem + OFF_P0);
  float*    Val    = (float*)(smem + OFF_P0);
  _Float16* EX     = (_Float16*)(smem + OFF_P1);
  _Float16* OUTF   = (_Float16*)(smem + OFF_P1);
  _Float16* Qs     = (_Float16*)(smem + OFF_P2);
  _Float16* Vt     = (_Float16*)(smem + OFF_P2);
  _Float16* Ks     = (_Float16*)(smem + OFF_P3);
  _Float16* ATT    = (_Float16*)(smem + OFF_P3);
  float*    logits = (float*)(smem + OFF_LOGITS);  // [32][4]
  float*    alpha  = (float*)(smem + OFF_ALPHA);   // [32][4]

  const int v    = blockIdx.x;
  const int tid  = threadIdx.x;
  const int lane = tid & 31;
  const int wave = tid >> 5;
  const int dg   = deg_arr[v];
  if (dg <= 0) return;               // uniform per block
  const int st   = start[v];

  const int col = lane & 15;
  const int mb  = (lane >> 4) << 3;  // 0 or 8 (row offset inside a 16-row tile)

  // ---- phase 1: gather cat(x[v], x[e1]) into A0, 128-bit copies ----
  {
    const int row = tid >> 3;        // 0..31
    const int seg = tid & 7;         // 8 segments of 32 halves (64B)
    uint4* dst = (uint4*)(A0 + row * PA + seg * 32);
    if (row < dg) {
      const uint4* src = (seg < 4)
          ? (const uint4*)(xh + (size_t)v * DIN + seg * 32)
          : (const uint4*)(xh + (size_t)e1[st + row] * DIN + (seg - 4) * 32);
#pragma unroll
      for (int i = 0; i < 4; ++i) dst[i] = src[i];
    } else {
      const uint4 z = {0u, 0u, 0u, 0u};
#pragma unroll
      for (int i = 0; i < 4; ++i) dst[i] = z;
    }
  }
  __syncthreads();

  // ---- phase 2: EX = gelu(A0 @ Wc^T + bc); 2 M-tiles share each B load ----
  for (int nt = wave; nt < 16; nt += 8) {
    v8f acc0 = {}, acc1 = {};
#pragma unroll
    for (int kt = 0; kt < 8; ++kt) {
      v16h b  = load_b16(WcH + (size_t)(nt * 16) * 256 + kt * 32, 256, lane);
      v16h a0 = load_a16(A0 + kt * 32, PA, lane);
      v16h a1 = load_a16(A0 + 16 * PA + kt * 32, PA, lane);
      acc0 = wmma_f16(a0, b, acc0);
      acc1 = wmma_f16(a1, b, acc1);
    }
    const int n = nt * 16 + col;
    const float bn = bc[n];
#pragma unroll
    for (int j = 0; j < 8; ++j) {
      EX[(mb + j) * PA + n]      = (_Float16)gelu_exact(acc0[j] + bn);
      EX[(mb + 16 + j) * PA + n] = (_Float16)gelu_exact(acc1[j] + bn);
    }
  }
  __syncthreads();

  // ---- phase 3: Q = EX@Wq^T+bq ; K = EX@Wk^T+bk ----
  for (int c = wave; c < 32; c += 8) {
    const int proj = c >> 4, nt = c & 15;
    const _Float16* W   = proj ? WkH : WqH;
    const float*    bia = proj ? bk  : bq;
    _Float16*       D   = proj ? Ks  : Qs;
    v8f acc0 = {}, acc1 = {};
#pragma unroll
    for (int kt = 0; kt < 8; ++kt) {
      v16h b  = load_b16(W + (size_t)(nt * 16) * 256 + kt * 32, 256, lane);
      v16h a0 = load_a16(EX + kt * 32, PA, lane);
      v16h a1 = load_a16(EX + 16 * PA + kt * 32, PA, lane);
      acc0 = wmma_f16(a0, b, acc0);
      acc1 = wmma_f16(a1, b, acc1);
    }
    const int n = nt * 16 + col;
    const float bn = bia[n];
#pragma unroll
    for (int j = 0; j < 8; ++j) {
      D[(mb + j) * PA + n]      = (_Float16)(acc0[j] + bn);
      D[(mb + 16 + j) * PA + n] = (_Float16)(acc1[j] + bn);
    }
  }
  __syncthreads();

  // ---- phase 4: per-head scores = (Qh @ Kh^T) * 1/8, key-padding mask ----
  {
    const int h = wave >> 1, nt = wave & 1;      // 8 combos, one per wave
    v8f acc0 = {}, acc1 = {};
#pragma unroll
    for (int kt = 0; kt < 2; ++kt) {
      v16h b  = load_b16(Ks + nt * 16 * PA + h * DH + kt * 32, PA, lane);
      v16h a0 = load_a16(Qs + h * DH + kt * 32, PA, lane);
      v16h a1 = load_a16(Qs + 16 * PA + h * DH + kt * 32, PA, lane);
      acc0 = wmma_f16(a0, b, acc0);
      acc1 = wmma_f16(a1, b, acc1);
    }
    const int key = nt * 16 + col;
    const bool ok = key < dg;
#pragma unroll
    for (int j = 0; j < 8; ++j) {
      S[h * 1024 + (mb + j) * 32 + key]      = ok ? acc0[j] * 0.125f : NEGV;
      S[h * 1024 + (mb + 16 + j) * 32 + key] = ok ? acc1[j] * 0.125f : NEGV;
    }
  }
  __syncthreads();

  // ---- phase 5: V^T = (EX@Wv^T+bv)^T ; softmax(S)->ATT ; zero logits ----
  for (int nt = wave; nt < 16; nt += 8) {
    v8f acc0 = {}, acc1 = {};
#pragma unroll
    for (int kt = 0; kt < 8; ++kt) {
      v16h b  = load_b16(WvH + (size_t)(nt * 16) * 256 + kt * 32, 256, lane);
      v16h a0 = load_a16(EX + kt * 32, PA, lane);
      v16h a1 = load_a16(EX + 16 * PA + kt * 32, PA, lane);
      acc0 = wmma_f16(a0, b, acc0);
      acc1 = wmma_f16(a1, b, acc1);
    }
    const int n = nt * 16 + col;
    const float bn = bv[n];
#pragma unroll
    for (int j = 0; j < 8; ++j) {
      Vt[n * PV + mb + j]      = (_Float16)(acc0[j] + bn);   // [chan][row]
      Vt[n * PV + 16 + mb + j] = (_Float16)(acc1[j] + bn);
    }
  }
  if (tid < 128) {                       // 4 heads x 32 query rows
    const int h = tid >> 5, q = tid & 31;
    const float* sr = S + h * 1024 + q * 32;
    float mx = sr[0];
#pragma unroll
    for (int k = 1; k < 32; ++k) mx = fmaxf(mx, sr[k]);
    float e[32]; float sum = 0.0f;
#pragma unroll
    for (int k = 0; k < 32; ++k) { e[k] = __expf(sr[k] - mx); sum += e[k]; }
    const float inv = 1.0f / sum;
    _Float16* ar = ATT + h * 32 * PT + q * PT;
#pragma unroll
    for (int k = 0; k < 32; ++k) ar[k] = (_Float16)(e[k] * inv);
  } else {                               // zero the 128 logit accumulators
    logits[tid - 128] = 0.0f;
  }
  __syncthreads();

  // ---- phase 6: OUTF = ATT @ V  (per head, K=32 -> single WMMA step) ----
  for (int c = wave; c < 16; c += 8) {
    const int h = c >> 2, nt = c & 3;
    v16h b  = load_b16(Vt + (size_t)(h * DH + nt * 16) * PV, PV, lane);
    v16h a0 = load_a16(ATT + h * 32 * PT, PT, lane);
    v16h a1 = load_a16(ATT + h * 32 * PT + 16 * PT, PT, lane);
    v8f acc0 = {}, acc1 = {};
    acc0 = wmma_f16(a0, b, acc0);
    acc1 = wmma_f16(a1, b, acc1);
    const int n = h * DH + nt * 16 + col;
#pragma unroll
    for (int j = 0; j < 8; ++j) {
      OUTF[(mb + j) * PA + n]      = (_Float16)acc0[j];
      OUTF[(mb + 16 + j) * PA + n] = (_Float16)acc1[j];
    }
  }
  __syncthreads();

  // ---- phase 7: gelu(OUTF @ Wo^T + bo) -> value half + gate-logit mean ----
  for (int nt = wave; nt < 16; nt += 8) {
    v8f acc0 = {}, acc1 = {};
#pragma unroll
    for (int kt = 0; kt < 8; ++kt) {
      v16h b  = load_b16(WoH + (size_t)(nt * 16) * 256 + kt * 32, 256, lane);
      v16h a0 = load_a16(OUTF + kt * 32, PA, lane);
      v16h a1 = load_a16(OUTF + 16 * PA + kt * 32, PA, lane);
      acc0 = wmma_f16(a0, b, acc0);
      acc1 = wmma_f16(a1, b, acc1);
    }
    const int n = nt * 16 + col;
    const float bn = bo[n];
    const int h = n >> 6, half = (n >> 5) & 1, cc = n & 31;
#pragma unroll
    for (int j = 0; j < 8; ++j) {
      const float g0 = gelu_exact(acc0[j] + bn);
      const float g1 = gelu_exact(acc1[j] + bn);
      if (half) {
        atomicAdd(&logits[(mb + j) * 4 + h],      g0 * (1.0f / 32.0f));
        atomicAdd(&logits[(mb + 16 + j) * 4 + h], g1 * (1.0f / 32.0f));
      } else {
        Val[(mb + j) * 128 + h * 32 + cc]      = g0;
        Val[(mb + 16 + j) * 128 + h * 32 + cc] = g1;
      }
    }
  }
  __syncthreads();

  // ---- phase 8: segment softmax over this node's deg rows (e0-segment) ----
  if (tid < HEADS) {
    const int h = tid;
    float mx = -3.402823466e38f;
    for (int r = 0; r < dg; ++r) mx = fmaxf(mx, logits[r * 4 + h]);
    float sum = 0.0f;
    for (int r = 0; r < dg; ++r) {
      const float e = __expf(logits[r * 4 + h] - mx);
      alpha[r * 4 + h] = e;
      sum += e;
    }
    const float inv = 1.0f / sum;
    for (int r = 0; r < dg; ++r) alpha[r * 4 + h] *= inv;
  }
  __syncthreads();

  // ---- phase 9: scatter contrib = value * alpha to out[e1] (f32 atomics) ----
  {
    const int row = tid >> 3, c0 = (tid & 7) * 16;
    if (row < dg) {
      const int dstn = e1[st + row];
      float* op = out + (size_t)dstn * 128;
#pragma unroll
      for (int cc = 0; cc < 16; ++cc) {
        const int c = c0 + cc;
        atomicAdd(op + c, Val[row * 128 + c] * alpha[row * 4 + (c >> 5)]);
      }
    }
  }
}

// ------------------------------- host launch -------------------------------

extern "C" void kernel_launch(void* const* d_in, const int* in_sizes, int n_in,
                              void* d_out, int out_size, void* d_ws, size_t ws_size,
                              hipStream_t stream) {
  const float* x  = (const float*)d_in[0];
  const float* Wc = (const float*)d_in[1];
  const float* bc = (const float*)d_in[2];
  const float* Wq = (const float*)d_in[3];
  const float* bq = (const float*)d_in[4];
  const float* Wk = (const float*)d_in[5];
  const float* bk = (const float*)d_in[6];
  const float* Wv = (const float*)d_in[7];
  const float* bv = (const float*)d_in[8];
  const float* Wo = (const float*)d_in[9];
  const float* bo = (const float*)d_in[10];
  const int*   e0  = (const int*)d_in[11];
  const int*   e1  = (const int*)d_in[12];
  const int*   pos = (const int*)d_in[13];
  const int m = in_sizes[11];

  // workspace layout (f16 casts of x + weights, then start/deg)
  char* ws = (char*)d_ws;
  _Float16* xh  = (_Float16*)ws;
  _Float16* WcH = xh  + (size_t)NODES * DIN;
  _Float16* WqH = WcH + DOUT * DOUT;
  _Float16* WkH = WqH + DOUT * DOUT;
  _Float16* WvH = WkH + DOUT * DOUT;
  _Float16* WoH = WvH + DOUT * DOUT;
  int* start = (int*)(WoH + DOUT * DOUT);
  int* deg   = start + NODES;

  const int T = 256;
  cvt_f32_f16<<<(NODES * DIN + T - 1) / T, T, 0, stream>>>(x,  xh,  NODES * DIN);
  cvt_f32_f16<<<(DOUT * DOUT + T - 1) / T, T, 0, stream>>>(Wc, WcH, DOUT * DOUT);
  cvt_f32_f16<<<(DOUT * DOUT + T - 1) / T, T, 0, stream>>>(Wq, WqH, DOUT * DOUT);
  cvt_f32_f16<<<(DOUT * DOUT + T - 1) / T, T, 0, stream>>>(Wk, WkH, DOUT * DOUT);
  cvt_f32_f16<<<(DOUT * DOUT + T - 1) / T, T, 0, stream>>>(Wv, WvH, DOUT * DOUT);
  cvt_f32_f16<<<(DOUT * DOUT + T - 1) / T, T, 0, stream>>>(Wo, WoH, DOUT * DOUT);
  zero_i32<<<(2 * NODES + T - 1) / T, T, 0, stream>>>(start, 2 * NODES);
  zero_f32<<<(NODES * 128 + T - 1) / T, T, 0, stream>>>((float*)d_out, NODES * 128);
  prep_edges<<<(m + T - 1) / T, T, 0, stream>>>(e0, pos, start, deg, m);

  node_attn_kernel<<<NODES, 256, SMEM_TOTAL, stream>>>(xh, WcH, WqH, WkH, WvH, WoH,
                                                       bc, bq, bk, bv, bo,
                                                       e1, start, deg, (float*)d_out);
  (void)n_in; (void)out_size; (void)ws_size;
}
